// RNNT_17274358465052
// MI455X (gfx1250) — matrix operand order repaired and checked by reading, not compiled, not run on hardware
//
#include <hip/hip_runtime.h>

// ---------------------------------------------------------------------------
// Types for CDNA5 WMMA (wave32, 16x16x32 bf16 -> f32 accumulate)
// ---------------------------------------------------------------------------
typedef __attribute__((ext_vector_type(16))) __bf16 v16bf;
typedef __attribute__((ext_vector_type(8)))  float  v8f;

union BfPack {
  uint4 q[2];
  unsigned u32[8];
  v16bf v;
  unsigned short us[16];
};

__device__ __forceinline__ unsigned short f2bf(float f) {
  union { float f; unsigned u; } x;
  x.f = f;
  unsigned r = x.u + 0x7FFFu + ((x.u >> 16) & 1u);   // round-to-nearest-even
  return (unsigned short)(r >> 16);
}

// Packed pair conversion: lo half = bf16(a), hi half = bf16(b)
__device__ __forceinline__ unsigned pk2bf(float a, float b) {
#if __has_builtin(__builtin_amdgcn_cvt_pk_bf16_f32)
  auto r = __builtin_amdgcn_cvt_pk_bf16_f32(a, b);
  unsigned u;
  __builtin_memcpy(&u, &r, 4);
  return u;
#else
  return (unsigned)f2bf(a) | ((unsigned)f2bf(b) << 16);
#endif
}

__device__ __forceinline__ v8f wmma_bf16(v16bf a, v16bf b, v8f c) {
  return __builtin_amdgcn_wmma_f32_16x16x32_bf16(false, a, false, b, (short)0, c,
                                                 false, false);
}

__device__ __forceinline__ void fill16(BfPack& p, float4 a, float4 b, float4 c, float4 d) {
  p.u32[0] = pk2bf(a.x, a.y); p.u32[1] = pk2bf(a.z, a.w);
  p.u32[2] = pk2bf(b.x, b.y); p.u32[3] = pk2bf(b.z, b.w);
  p.u32[4] = pk2bf(c.x, c.y); p.u32[5] = pk2bf(c.z, c.w);
  p.u32[6] = pk2bf(d.x, d.y); p.u32[7] = pk2bf(d.z, d.w);
}

// B pre-packed: [N/16][ceil(K/32)][lane 0..31][16 halves] -> one 32B load per lane
__device__ __forceinline__ BfPack loadB(const unsigned short* __restrict__ Bp,
                                        size_t nt16, int Ktiles, int kt, int lane) {
  const unsigned short* bp = Bp + ((nt16 * (size_t)Ktiles + kt) * 32 + lane) * 16;
  BfPack p;
  p.q[0] = *(const uint4*)(bp);
  p.q[1] = *(const uint4*)(bp + 8);
  return p;
}

// CDNA5 async copy: one 16B chunk global -> LDS, tracked by ASYNCcnt
__device__ __forceinline__ void async_g2l_b128(unsigned ldsOff, const void* gaddr) {
  asm volatile("global_load_async_to_lds_b128 %0, %1, off"
               :: "v"(ldsOff), "v"(gaddr) : "memory");
}
__device__ __forceinline__ void wait_asynccnt0() {
  asm volatile("s_wait_asynccnt 0x0" ::: "memory");
}

__device__ __forceinline__ float sigf(float x) { return 1.0f / (1.0f + expf(-x)); }

// ---------------------------------------------------------------------------
// Pack fp32 [K,N] weight into WMMA-B bf16 layout (zero-pads K to mult of 32).
// Element (nt,kt,lane,e): srcK = kt*32 + (e&7) + (lane>=16 ? 8 : 0) + (e>=8 ? 16 : 0), srcN = nt*16 + lane%16
// ---------------------------------------------------------------------------
__global__ void pack_b_kernel(const float* __restrict__ B, unsigned short* __restrict__ out,
                              int K, int N) {
  int Ktiles = (K + 31) >> 5;
  size_t total = (size_t)(N >> 4) * (size_t)Ktiles * 512;
  size_t idx = (size_t)blockIdx.x * blockDim.x + threadIdx.x;
  if (idx >= total) return;
  int e    = (int)(idx & 15);
  int lane = (int)((idx >> 4) & 31);
  size_t rest = idx >> 9;
  int kt = (int)(rest % Ktiles);
  int nt = (int)(rest / Ktiles);
  int srcK = (kt << 5) + (e & 7) + ((lane >= 16) ? 8 : 0) + ((e >= 8) ? 16 : 0);
  int srcN = (nt << 4) + (lane & 15);
  out[idx] = (srcK < K) ? f2bf(B[(size_t)srcK * N + srcN]) : (unsigned short)0;
}

// ---------------------------------------------------------------------------
// BatchNorm (inference, moving stats), per-feature
// ---------------------------------------------------------------------------
__global__ void bn_kernel(const float* __restrict__ x, const float* __restrict__ g,
                          const float* __restrict__ b, const float* __restrict__ m,
                          const float* __restrict__ v, float* __restrict__ out,
                          int total, int F) {
  int idx = blockIdx.x * blockDim.x + threadIdx.x;
  if (idx >= total) return;
  int f = idx % F;
  out[idx] = (x[idx] - m[f]) * rsqrtf(v[f] + 1e-3f) * g[f] + b[f];
}

// ---------------------------------------------------------------------------
// Embedding gather
// ---------------------------------------------------------------------------
__global__ void emb_kernel(const int* __restrict__ pre, const float* __restrict__ emb,
                           float* __restrict__ out, int rows, int E) {
  int idx = blockIdx.x * blockDim.x + threadIdx.x;
  if (idx >= rows * E) return;
  int r = idx / E, e = idx - r * E;
  out[idx] = emb[(size_t)pre[r] * E + e];
}

// ---------------------------------------------------------------------------
// LayerNorm, one wave (32 lanes) per row, in place
// ---------------------------------------------------------------------------
__global__ void ln_kernel(float* __restrict__ x, const float* __restrict__ g,
                          const float* __restrict__ b, int rows, int len) {
  int wave = blockIdx.x * (blockDim.x >> 5) + (threadIdx.x >> 5);
  int lane = threadIdx.x & 31;
  if (wave >= rows) return;
  float* r = x + (size_t)wave * len;
  float s = 0.f, s2 = 0.f;
  for (int i = lane; i < len; i += 32) { float v = r[i]; s += v; s2 += v * v; }
  for (int o = 16; o; o >>= 1) { s += __shfl_xor(s, o, 32); s2 += __shfl_xor(s2, o, 32); }
  float mean = s / len;
  float var  = s2 / len - mean * mean;
  float rstd = rsqrtf(var + 1e-3f);
  for (int i = lane; i < len; i += 32) r[i] = (r[i] - mean) * rstd * g[i] + b[i];
}

// ---------------------------------------------------------------------------
// Block-cooperative GEMM: C[M,N] = act(A @ Bpacked + bias).
// Block = 256 threads = 8 waves covering 8 consecutive M-tiles (128 rows) that
// SHARE one 64-column B strip. Per k-tile the 4 B WMMA-tiles (4 KB) are staged
// into LDS with global_load_async_to_lds_b128 (one 16B chunk per thread),
// synchronized via s_wait_asynccnt + barrier -> 8x B reuse from LDS.
// Requires M % 128 == 0, N % 64 == 0; K arbitrary (tail path zero-pads).
// ---------------------------------------------------------------------------
__global__ void __launch_bounds__(256)
gemm_kernel(const float* __restrict__ A, const unsigned short* __restrict__ Bp,
            const float* __restrict__ bias, float* __restrict__ C,
            int M, int N, int K, int act) {
  __shared__ unsigned short sB[4 * 512];   // 4 tiles x (32 lanes x 16 halves)
  int tid  = threadIdx.x;
  int wave = tid >> 5;
  int lane = tid & 31;
  int ng = N >> 6;
  int mb = blockIdx.x / ng;
  int nb = blockIdx.x - mb * ng;
  int Ktiles = (K + 31) >> 5;

  // async-stage assignment: thread -> (tile ti, 16B chunk)
  int ti    = tid >> 6;     // 0..3
  int chunk = tid & 63;     // 0..63
  const unsigned short* src0 =
      Bp + ((size_t)(nb * 4 + ti) * Ktiles) * 512 + (size_t)chunk * 8;
  unsigned ldsOff = (unsigned)(size_t)(&sB[ti * 512 + chunk * 8]);

  int mt  = (mb << 3) + wave;
  int row = (mt << 4) + (lane & 15);
  int klo = (lane >> 4) << 3;
  v8f acc[4] = {};

  for (int kt = 0; kt < Ktiles; ++kt) {
    async_g2l_b128(ldsOff, src0 + (size_t)kt * 512);
    // A fragment (overlaps with async copy)
    int kb = (kt << 5) + klo;
    BfPack pa;
    if (((kt + 1) << 5) <= K) {
      const float* ap = A + (size_t)row * K + kb;
      fill16(pa, *(const float4*)(ap), *(const float4*)(ap + 4),
                 *(const float4*)(ap + 16), *(const float4*)(ap + 20));
    } else {
#pragma unroll
      for (int e = 0; e < 16; ++e) {
        int k = kb + e + ((e >= 8) ? 8 : 0);
        pa.us[e] = (k < K) ? f2bf(A[(size_t)row * K + k]) : (unsigned short)0;
      }
    }
    wait_asynccnt0();
    __syncthreads();
#pragma unroll
    for (int i = 0; i < 4; ++i) {
      const unsigned short* sp = &sB[i * 512 + lane * 16];
      BfPack pb;
      pb.q[0] = *(const uint4*)(sp);
      pb.q[1] = *(const uint4*)(sp + 8);
      acc[i] = wmma_bf16(pa.v, pb.v, acc[i]);
    }
    __syncthreads();
  }

  int col0  = (nb << 6) + (lane & 15);
  int rbase = (mt << 4) + ((lane >> 4) << 3);
#pragma unroll
  for (int i = 0; i < 4; ++i) {
    int col = col0 + (i << 4);
    float bv = bias[col];
#pragma unroll
    for (int r = 0; r < 8; ++r) {
      float v = acc[i][r] + bv;
      if (act) v = tanhf(v);
      C[(size_t)(rbase + r) * N + col] = v;
    }
  }
}

// ---------------------------------------------------------------------------
// Joint stage 1: C = tanh((enc[b,t,:] + pred[b,u,:]) @ j1 + j1_b)
// Same block-cooperative LDS scheme; A rows synthesized on the fly
// (never materializes the [B,T',U,D] tensor).
// ---------------------------------------------------------------------------
__global__ void __launch_bounds__(256)
joint1_kernel(const float* __restrict__ enc, const float* __restrict__ pred,
              const unsigned short* __restrict__ Bp, const float* __restrict__ bias,
              float* __restrict__ C, int Tp, int U, int D, int N) {
  __shared__ unsigned short sB[4 * 512];
  int tid  = threadIdx.x;
  int wave = tid >> 5;
  int lane = tid & 31;
  int ng = N >> 6;
  int mb = blockIdx.x / ng;
  int nb = blockIdx.x - mb * ng;
  int Ktiles = D >> 5;

  int ti    = tid >> 6;
  int chunk = tid & 63;
  const unsigned short* src0 =
      Bp + ((size_t)(nb * 4 + ti) * Ktiles) * 512 + (size_t)chunk * 8;
  unsigned ldsOff = (unsigned)(size_t)(&sB[ti * 512 + chunk * 8]);

  int mt  = (mb << 3) + wave;
  int row = (mt << 4) + (lane & 15);
  int bb  = row / (Tp * U);
  int rem = row - bb * Tp * U;
  int tt  = rem / U;
  int uu  = rem - tt * U;
  const float* er = enc  + ((size_t)bb * Tp + tt) * D;
  const float* pr = pred + ((size_t)bb * U  + uu) * D;
  int klo = (lane >> 4) << 3;
  v8f acc[4] = {};

  for (int kt = 0; kt < Ktiles; ++kt) {
    async_g2l_b128(ldsOff, src0 + (size_t)kt * 512);
    int kb = (kt << 5) + klo;
    float4 e0 = *(const float4*)(er + kb),      e1 = *(const float4*)(er + kb + 4);
    float4 e2 = *(const float4*)(er + kb + 16), e3 = *(const float4*)(er + kb + 20);
    float4 p0 = *(const float4*)(pr + kb),      p1 = *(const float4*)(pr + kb + 4);
    float4 p2 = *(const float4*)(pr + kb + 16), p3 = *(const float4*)(pr + kb + 20);
    float4 s0 = {e0.x + p0.x, e0.y + p0.y, e0.z + p0.z, e0.w + p0.w};
    float4 s1 = {e1.x + p1.x, e1.y + p1.y, e1.z + p1.z, e1.w + p1.w};
    float4 s2 = {e2.x + p2.x, e2.y + p2.y, e2.z + p2.z, e2.w + p2.w};
    float4 s3 = {e3.x + p3.x, e3.y + p3.y, e3.z + p3.z, e3.w + p3.w};
    BfPack pa;
    fill16(pa, s0, s1, s2, s3);
    wait_asynccnt0();
    __syncthreads();
#pragma unroll
    for (int i = 0; i < 4; ++i) {
      const unsigned short* sp = &sB[i * 512 + lane * 16];
      BfPack pb;
      pb.q[0] = *(const uint4*)(sp);
      pb.q[1] = *(const uint4*)(sp + 8);
      acc[i] = wmma_bf16(pa.v, pb.v, acc[i]);
    }
    __syncthreads();
  }

  int col0  = (nb << 6) + (lane & 15);
  int rbase = (mt << 4) + ((lane >> 4) << 3);
#pragma unroll
  for (int i = 0; i < 4; ++i) {
    int col = col0 + (i << 4);
    float bv = bias[col];
#pragma unroll
    for (int r = 0; r < 8; ++r)
      C[(size_t)(rbase + r) * N + col] = tanhf(acc[i][r] + bv);
  }
}

// ---------------------------------------------------------------------------
// One fused LSTM timestep: z = xz[:,t,:] + h_{t-1} @ Wh ; gate math ; update c ;
// write h_t into the sequence buffer (which doubles as h_{t-1} storage).
// Each wave owns 16 hidden columns with 4 accumulators (gates i,f,g,o) sharing
// one A-fragment; B=2 -> rows 0..1 of the 16-row tile are live.
// Launch: 64 waves (16 blocks x 128 threads) for H=1024.
// ---------------------------------------------------------------------------
__global__ void lstm_step_kernel(const float* __restrict__ xz,   // [2*T, 4H]
                                 float* __restrict__ hseq,       // [2*T, H]
                                 const unsigned short* __restrict__ Whp, // packed, Ntot=4H
                                 float* __restrict__ c,          // [2*H]
                                 int T, int H, int t, int first) {
  int wave = blockIdx.x * (blockDim.x >> 5) + (threadIdx.x >> 5);
  int lane = threadIdx.x & 31;
  int ntiles = H >> 4;
  if (wave >= ntiles) return;
  int Ktiles = H >> 5;
  v8f acc[4] = {};
  if (!first) {
    int bidx = lane & 15;
    bool rowvalid = bidx < 2;
    int klo = (lane >> 4) << 3;
    for (int kt = 0; kt < Ktiles; ++kt) {
      int kb = (kt << 5) + klo;
      BfPack pa;
      if (rowvalid) {
        const float* hp = hseq + ((size_t)bidx * T + (t - 1)) * H + kb;
        fill16(pa, *(const float4*)(hp), *(const float4*)(hp + 4),
                   *(const float4*)(hp + 16), *(const float4*)(hp + 20));
      } else {
        pa.q[0] = make_uint4(0, 0, 0, 0);
        pa.q[1] = make_uint4(0, 0, 0, 0);
      }
#pragma unroll
      for (int g = 0; g < 4; ++g) {
        BfPack pb = loadB(Whp, (size_t)(g * ntiles + wave), Ktiles, kt, lane);
        acc[g] = wmma_bf16(pa.v, pb.v, acc[g]);
      }
    }
  }
  if (lane < 16) {
    int j = (wave << 4) + lane;
#pragma unroll
    for (int b = 0; b < 2; ++b) {
      size_t zrow = ((size_t)b * T + t) * (size_t)(4 * H);
      float zi = acc[0][b] + xz[zrow + 0 * H + j];
      float zf = acc[1][b] + xz[zrow + 1 * H + j];
      float zg = acc[2][b] + xz[zrow + 2 * H + j];
      float zo = acc[3][b] + xz[zrow + 3 * H + j];
      float cc = sigf(zf) * c[b * H + j] + sigf(zi) * tanhf(zg);
      c[b * H + j] = cc;
      hseq[((size_t)b * T + t) * H + j] = sigf(zo) * tanhf(cc);
    }
  }
}

// ---------------------------------------------------------------------------
// Host orchestration
// ---------------------------------------------------------------------------
extern "C" void kernel_launch(void* const* d_in, const int* in_sizes, int n_in,
                              void* d_out, int out_size, void* d_ws, size_t ws_size,
                              hipStream_t stream) {
  (void)in_sizes; (void)n_in; (void)out_size; (void)ws_size;

  const int B = 2, T = 384, F = 80, U = 128, H = 1024, Pe = 512;
  const int E = 1024, Pp = 1024, J = 640, Tp = 192, D = 1024;

  const float* enc_in = (const float*)d_in[0];
  const int*   pre    = (const int*)d_in[1];
  const float* bn_g = (const float*)d_in[2];
  const float* bn_b = (const float*)d_in[3];
  const float* bn_m = (const float*)d_in[4];
  const float* bn_v = (const float*)d_in[5];
  const float* eWx[3]; const float* eWh[3]; const float* eB[3];
  const float* eWd[3]; const float* eBd[3]; const float* eG[3]; const float* eBe[3];
  for (int i = 0; i < 3; ++i) {
    int base = 6 + 7 * i;
    eWx[i] = (const float*)d_in[base + 0];
    eWh[i] = (const float*)d_in[base + 1];
    eB[i]  = (const float*)d_in[base + 2];
    eWd[i] = (const float*)d_in[base + 3];
    eBd[i] = (const float*)d_in[base + 4];
    eG[i]  = (const float*)d_in[base + 5];
    eBe[i] = (const float*)d_in[base + 6];
  }
  const float* emb = (const float*)d_in[27];
  const float* pWx[2]; const float* pWh[2]; const float* pB[2];
  const float* pWd[2]; const float* pBd[2]; const float* pG[2]; const float* pBe[2];
  for (int i = 0; i < 2; ++i) {
    int base = 28 + 7 * i;
    pWx[i] = (const float*)d_in[base + 0];
    pWh[i] = (const float*)d_in[base + 1];
    pB[i]  = (const float*)d_in[base + 2];
    pWd[i] = (const float*)d_in[base + 3];
    pBd[i] = (const float*)d_in[base + 4];
    pG[i]  = (const float*)d_in[base + 5];
    pBe[i] = (const float*)d_in[base + 6];
  }
  const float* j1W = (const float*)d_in[42];
  const float* j1b = (const float*)d_in[43];
  const float* j2W = (const float*)d_in[44];
  const float* j2b = (const float*)d_in[45];

  // ---- workspace bump allocator ----
  char* wp = (char*)d_ws;
  auto allocB = [&](size_t bytes) -> void* {
    void* p = (void*)wp;
    wp += (bytes + 255) & ~(size_t)255;
    return p;
  };
  auto packElems = [](int K, int N) -> size_t {
    return (size_t)((K + 31) / 32) * 32 * (size_t)N;
  };

  unsigned short* eWxp[3]; unsigned short* eWhp[3]; unsigned short* eWdp[3];
  const int eWxK[3] = {F, Pe, Pe};
  for (int i = 0; i < 3; ++i) {
    eWxp[i] = (unsigned short*)allocB(packElems(eWxK[i], 4 * H) * 2);
    eWhp[i] = (unsigned short*)allocB(packElems(H, 4 * H) * 2);
    eWdp[i] = (unsigned short*)allocB(packElems(H, Pe) * 2);
  }
  unsigned short* pWxp[2]; unsigned short* pWhp[2]; unsigned short* pWdp[2];
  const int pWxK[2] = {E, Pp};
  for (int i = 0; i < 2; ++i) {
    pWxp[i] = (unsigned short*)allocB(packElems(pWxK[i], 4 * H) * 2);
    pWhp[i] = (unsigned short*)allocB(packElems(H, 4 * H) * 2);
    pWdp[i] = (unsigned short*)allocB(packElems(H, Pp) * 2);
  }
  unsigned short* j1p = (unsigned short*)allocB(packElems(D, J) * 2);
  unsigned short* j2p = (unsigned short*)allocB(packElems(J, 1024) * 2);

  float* bnout  = (float*)allocB((size_t)B * T * F * 4);
  float* xz     = (float*)allocB((size_t)B * T * 4 * H * 4);
  float* hseq   = (float*)allocB((size_t)B * T * H * 4);
  float* proj   = (float*)allocB((size_t)B * T * Pe * 4);
  float* encbuf = (float*)allocB((size_t)B * T * Pe * 4);   // == [B,Tp,D] view
  float* predy  = (float*)allocB((size_t)B * U * E * 4);
  float* pbuf0  = (float*)allocB((size_t)B * U * Pp * 4);
  float* pbuf1  = (float*)allocB((size_t)B * U * Pp * 4);
  float* cbuf   = (float*)allocB((size_t)B * H * 4);
  float* jointh = (float*)allocB((size_t)B * Tp * U * J * 4);

  // ---- launch helpers ----
  auto pack = [&](const float* W, unsigned short* dst, int K, int N) {
    size_t total = packElems(K, N);
    int blocks = (int)((total + 255) / 256);
    pack_b_kernel<<<blocks, 256, 0, stream>>>(W, dst, K, N);
  };
  auto gemm = [&](const float* A, const unsigned short* Bp, const float* bias,
                  float* C, int M, int N, int K, int act) {
    int blocks = (M >> 7) * (N >> 6);    // M % 128 == 0, N % 64 == 0
    gemm_kernel<<<blocks, 256, 0, stream>>>(A, Bp, bias, C, M, N, K, act);
  };
  auto layernorm = [&](float* x, const float* g, const float* b, int rows, int len) {
    int blocks = (rows + 3) / 4;
    ln_kernel<<<blocks, 128, 0, stream>>>(x, g, b, rows, len);
  };
  auto lstm = [&](const float* xzp, float* hs, const unsigned short* Whpk, int Tloc) {
    hipMemsetAsync(cbuf, 0, (size_t)B * H * sizeof(float), stream);
    for (int t = 0; t < Tloc; ++t)
      lstm_step_kernel<<<16, 128, 0, stream>>>(xzp, hs, Whpk, cbuf, Tloc, H, t, t == 0);
  };

  // ---- pack all weights (bf16, WMMA-B layout) ----
  for (int i = 0; i < 3; ++i) {
    pack(eWx[i], eWxp[i], eWxK[i], 4 * H);
    pack(eWh[i], eWhp[i], H, 4 * H);
    pack(eWd[i], eWdp[i], H, Pe);
  }
  for (int i = 0; i < 2; ++i) {
    pack(pWx[i], pWxp[i], pWxK[i], 4 * H);
    pack(pWh[i], pWhp[i], H, 4 * H);
    pack(pWd[i], pWdp[i], H, Pp);
  }
  pack(j1W, j1p, D, J);
  pack(j2W, j2p, J, 1024);

  // ---- encoder ----
  {
    int total = B * T * F;
    bn_kernel<<<(total + 255) / 256, 256, 0, stream>>>(enc_in, bn_g, bn_b, bn_m, bn_v,
                                                       bnout, total, F);
  }
  gemm(bnout, eWxp[0], eB[0], xz, B * T, 4 * H, F, 0);
  lstm(xz, hseq, eWhp[0], T);
  gemm(hseq, eWdp[0], eBd[0], proj, B * T, Pe, H, 0);
  layernorm(proj, eG[0], eBe[0], B * T, Pe);

  gemm(proj, eWxp[1], eB[1], xz, B * T, 4 * H, Pe, 0);
  lstm(xz, hseq, eWhp[1], T);
  gemm(hseq, eWdp[1], eBd[1], proj, B * T, Pe, H, 0);
  layernorm(proj, eG[1], eBe[1], B * T, Pe);

  gemm(proj, eWxp[2], eB[2], xz, B * T, 4 * H, Pe, 0);
  lstm(xz, hseq, eWhp[2], T);
  gemm(hseq, eWdp[2], eBd[2], encbuf, B * T, Pe, H, 0);
  layernorm(encbuf, eG[2], eBe[2], B * T, Pe);
  // time reduction [2,384,512] -> [2,192,1024] is a pure view of encbuf

  // ---- prediction network ----
  emb_kernel<<<(B * U * E + 255) / 256, 256, 0, stream>>>(pre, emb, predy, B * U, E);
  gemm(predy, pWxp[0], pB[0], xz, B * U, 4 * H, E, 0);
  lstm(xz, hseq, pWhp[0], U);
  gemm(hseq, pWdp[0], pBd[0], pbuf0, B * U, Pp, H, 0);
  layernorm(pbuf0, pG[0], pBe[0], B * U, Pp);

  gemm(pbuf0, pWxp[1], pB[1], xz, B * U, 4 * H, Pp, 0);
  lstm(xz, hseq, pWhp[1], U);
  gemm(hseq, pWdp[1], pBd[1], pbuf1, B * U, Pp, H, 0);
  layernorm(pbuf1, pG[1], pBe[1], B * U, Pp);

  // ---- joint network ----
  {
    int M = B * Tp * U;
    int blocks = (M >> 7) * (J >> 6);
    joint1_kernel<<<blocks, 256, 0, stream>>>(encbuf, pbuf1, j1p, j1b, jointh,
                                              Tp, U, D, J);
  }
  gemm(jointh, j2p, j2b, (float*)d_out, B * Tp * U, 1024, J, 0);
}
